// SEMHSADahazeFormer_30863634989268
// MI455X (gfx1250) — matrix operand order repaired and checked by reading, not compiled, not run on hardware
//
#include <hip/hip_runtime.h>
#include <hip/hip_bf16.h>

typedef __attribute__((ext_vector_type(16))) _Float16 v16h;
typedef __attribute__((ext_vector_type(8)))  _Float16 v8h;
typedef __attribute__((ext_vector_type(8)))  float    v8f;

#define CDIM 384
#define NHEADS 12
#define HD 32
#define NTOK 64

// LDS strides (element units)
#define QS 40   // Qt/Kt       [64][40] f16
#define VS 72   // Vt          [32][72] f16 (transposed v)
#define AS 72   // At          [64][72] f16 (softmaxed attn)
#define PS 65   // attnS       [64][65] f32

// dynamic-LDS byte offsets (all multiples of 16)
#define OFF_BIAS   0        // 2700 f32  = 10800
#define OFF_QT     10800    // 2560 f16  = 5120
#define OFF_KT     15920    // 5120
#define OFF_VT     21040    // 2304 f16  = 4608
#define OFF_RED    25648    // 144 f32   = 576
#define OFF_QRAW   26224    // 2048 f32  = 8192   (phase-1 region)
#define OFF_KRAW   34416    // 8192  -> ends 42608
#define OFF_ATTNS  26224    // 4160 f32  = 16640  (phase-2 region, aliases phase-1)
#define OFF_AT     42864    // 4608 f16  = 9216 -> ends 52080
#define SMEM_BYTES 52096

union f8u { v8f v; float f[8]; };

__device__ __forceinline__ v16h load_fragA(const _Float16* base, int stride, int row, int kb, int hi) {
  // A 16x32 f16 layout: lane m=lane&15 holds K = {hi*8..+7} and {16+hi*8..+7}
  const _Float16* p = base + (long)row * stride + kb + hi * 8;
  union { v16h v; v8h h[2]; } u;
  u.h[0] = *(const v8h*)(p);
  u.h[1] = *(const v8h*)(p + 16);
  return u.v;
}

__device__ __forceinline__ v16h load_fragB(const _Float16* base, int stride, int row, int kb, int hi) {
  // B 32x16 f16 layout: lane n=lane&15 holds K = {hi*8..+7} and {16+hi*8..+7}
  const _Float16* p = base + (long)row * stride + kb + hi * 8;
  union { v16h v; v8h h[2]; } u;
  u.h[0] = *(const v8h*)(p);
  u.h[1] = *(const v8h*)(p + 16);
  return u.v;
}

__device__ __forceinline__ v8f wmma32(v16h a, v16h b, v8f c) {
  return __builtin_amdgcn_wmma_f32_16x16x32_f16(false, a, false, b, (short)0, c, false, false);
}

// ---------------------------------------------------------------------------
// K0a: pack [w_qk ; w_v] -> f16 Wqkv (1152x384 row-major), and bias vector
// ---------------------------------------------------------------------------
__global__ __launch_bounds__(256) void prep_qkv_w(
    const float* __restrict__ w_qk, const float* __restrict__ b_qk,
    const float* __restrict__ w_v,  const float* __restrict__ b_v,
    _Float16* __restrict__ Wq, float* __restrict__ bq)
{
  int idx = blockIdx.x * 256 + threadIdx.x;
  if (idx < 1152 * CDIM) {
    int o = idx / CDIM, c = idx - o * CDIM;
    float v = (o < 768) ? w_qk[idx] : w_v[(o - 768) * CDIM + c];
    Wq[idx] = (_Float16)v;
  }
  if (idx < 1152) bq[idx] = (idx < 768) ? b_qk[idx] : b_v[idx - 768];
}

// ---------------------------------------------------------------------------
// K0b: Wcomb = w_proj @ w_po  (384x384, f16) — folds the two output 1x1 convs
// ---------------------------------------------------------------------------
__global__ __launch_bounds__(256) void prep_comb_w(
    const float* __restrict__ w_proj, const float* __restrict__ w_po,
    _Float16* __restrict__ Wc)
{
  int idx = blockIdx.x * 256 + threadIdx.x;
  if (idx >= CDIM * CDIM) return;
  int o = idx / CDIM, c = idx - o * CDIM;
  float s = 0.f;
  for (int m = 0; m < CDIM; m++) s += w_proj[o * CDIM + m] * w_po[m * CDIM + c];
  Wc[idx] = (_Float16)s;
}

// ---------------------------------------------------------------------------
// K0c: transpose x (NCHW f32) -> X16 (window-token-major rows x channels, f16)
// Each thread: one float4 (4 pixels along W, same channel) -> 4 f16 stores.
// 4 consecutive pixels never cross an 8-wide window boundary.
// ---------------------------------------------------------------------------
__global__ __launch_bounds__(256) void prep_x(
    const float* __restrict__ x, _Float16* __restrict__ X16)
{
  int idx = blockIdx.x * 256 + threadIdx.x;           // 6291456 total
  const float4 v = ((const float4*)x)[idx];
  int t = idx;
  int w4 = t & 31;  t >>= 5;                           // wcol/4
  int h  = t & 127; t >>= 7;
  int c  = t % CDIM;
  int b  = t / CDIM;
  int wy = h >> 3, iy = h & 7;
  int wcol = w4 * 4;
  int wx = wcol >> 3, ixb = wcol & 7;
  int wi = (b * 16 + wy) * 16 + wx;
  long row0 = (long)wi * NTOK + iy * 8 + ixb;
  float vv[4] = {v.x, v.y, v.z, v.w};
#pragma unroll
  for (int j = 0; j < 4; j++)
    X16[(row0 + j) * CDIM + c] = (_Float16)vv[j];
}

// ---------------------------------------------------------------------------
// K1: relative-position bias table — bias depends only on (dy,dx) in 15x15
// ---------------------------------------------------------------------------
__global__ __launch_bounds__(128) void bias_kernel(
    const float* __restrict__ posw, const float* __restrict__ pscale,
    const float* __restrict__ mw1, const float* __restrict__ mb1,
    const float* __restrict__ mw2, const float* __restrict__ mb2,
    float* __restrict__ biasT)
{
  __shared__ float hbuf[128];
  int idx = blockIdx.x;                  // 0..224
  int dyi = idx / 15 - 7, dxi = idx % 15 - 7;
  float dx = dxi * (2.f / 7.f), dy = dyi * (2.f / 7.f);
  float sc = pscale[0];
  int j = threadIdx.x;                   // 128
  float s = mb1[j];
  for (int c = 0; c < CDIM; c++) {
    float relc = sc * (posw[c * 2 + 0] * dx + posw[c * 2 + 1] * dy);
    s += relc * mw1[j * CDIM + c];
  }
  hbuf[j] = 0.5f * s * (1.f + erff(s * 0.70710678118f));  // exact GELU
  __syncthreads();
  if (j < NHEADS) {
    float b = mb2[j];
    for (int q = 0; q < 128; q++) b += hbuf[q] * mw2[j * 128 + q];
    biasT[idx * NHEADS + j] = b;
  }
}

// ---------------------------------------------------------------------------
// K2: fused per-(window,head) QKV + PDSCA gating + attention
// grid (1024 windows, 12 heads), 192 threads (6 waves)
// QKV phase is LDS-free: A/B fragments loaded directly from X16/Wqkv (f16).
// ---------------------------------------------------------------------------
__global__ __launch_bounds__(192) void attn_kernel(
    const _Float16* __restrict__ X16, const _Float16* __restrict__ Wqkv,
    const float* __restrict__ bqkv, const float* __restrict__ log_temp,
    const float* __restrict__ biasTg,
    const float* __restrict__ pq_dw_w, const float* __restrict__ pq_dw_b,
    const float* __restrict__ pq_f1w, const float* __restrict__ pq_f1b,
    const float* __restrict__ pq_f2w, const float* __restrict__ pq_f2b,
    const float* __restrict__ pk_dw_w, const float* __restrict__ pk_dw_b,
    const float* __restrict__ pk_f1w, const float* __restrict__ pk_f1b,
    const float* __restrict__ pk_f2w, const float* __restrict__ pk_f2b,
    _Float16* __restrict__ O)
{
  extern __shared__ char smem[];
  const int w = blockIdx.x, head = blockIdx.y;
  const int tid = threadIdx.x, wave = tid >> 5, lane = tid & 31;
  const int ln = lane & 15, hi = lane >> 4;

  float*    biasT = (float*)(smem + OFF_BIAS);
  _Float16* Qt    = (_Float16*)(smem + OFF_QT);
  _Float16* Kt    = (_Float16*)(smem + OFF_KT);
  _Float16* Vt    = (_Float16*)(smem + OFF_VT);
  float*    red   = (float*)(smem + OFF_RED);
  float*    q_raw = (float*)(smem + OFF_QRAW);
  float*    k_raw = (float*)(smem + OFF_KRAW);
  float*    attnS = (float*)(smem + OFF_ATTNS);
  _Float16* At    = (_Float16*)(smem + OFF_AT);
  float *gq = red, *gk = red + 32, *zq = red + 64, *zk = red + 72,
        *cq = red + 80, *ck = red + 112;

  for (int i = tid; i < 225 * NHEADS; i += 192) biasT[i] = biasTg[i];
  const float temp = expf(log_temp[0]);

  // ---------- phase 1: QKV GEMM (64 tokens x 96 channels, K=384) ----------
  const int sel = wave >> 1;                                   // 0=q 1=k 2=v
  const int wrow = sel * CDIM + head * HD + (wave & 1) * 16 + ln;
  f8u acc[4];
  for (int mt = 0; mt < 4; mt++)
    for (int r = 0; r < 8; r++) acc[mt].f[r] = 0.f;

  for (int ks = 0; ks < 12; ks++) {
    int kb = ks * 32;
    v16h fb = load_fragB(Wqkv, CDIM, wrow, kb, hi);
#pragma unroll
    for (int mt = 0; mt < 4; mt++) {
      v16h fa = load_fragA(X16, CDIM, w * NTOK + mt * 16 + ln, kb, hi);
      acc[mt].v = wmma32(fa, fb, acc[mt].v);
    }
  }
  {
    float bv = bqkv[wrow];
#pragma unroll
    for (int mt = 0; mt < 4; mt++)
      for (int r = 0; r < 8; r++) {
        int token = mt * 16 + r + hi * 8;
        float val = acc[mt].f[r] + bv;
        int d = (wave & 1) * 16 + ln;
        if (sel == 0)      q_raw[token * HD + d] = val;
        else if (sel == 1) k_raw[token * HD + d] = val;
        else               Vt[d * VS + token] = (_Float16)val;
      }
  }
  __syncthreads();

  // ---------- phase 1b: PDSCA (SE channel branch) ----------
  if (tid < 64) {
    int d = tid & 31;
    const float* raw = (tid < 32) ? q_raw : k_raw;
    float s = 0.f;
    for (int n = 0; n < NTOK; n++) s += raw[n * HD + d];
    ((tid < 32) ? gq : gk)[d] = s * (1.f / 64.f);
  }
  __syncthreads();
  if (tid < 16) {
    int j = tid & 7;
    const float *g = (tid < 8) ? gq : gk;
    const float *f1w = (tid < 8) ? pq_f1w : pk_f1w;
    const float *f1b = (tid < 8) ? pq_f1b : pk_f1b;
    float s = f1b[j];
    for (int d = 0; d < HD; d++) s += f1w[j * HD + d] * g[d];
    ((tid < 8) ? zq : zk)[j] = fmaxf(s, 0.f);
  }
  __syncthreads();
  if (tid < 64) {
    int d = tid & 31;
    const float *z = (tid < 32) ? zq : zk;
    const float *f2w = (tid < 32) ? pq_f2w : pk_f2w;
    const float *f2b = (tid < 32) ? pq_f2b : pk_f2b;
    float s = f2b[d];
    for (int r = 0; r < 8; r++) s += f2w[d * 8 + r] * z[r];
    ((tid < 32) ? cq : ck)[d] = s;
  }
  __syncthreads();

  // ---------- phase 1c: depthwise 3x3 + sigmoid gate ----------
  for (int e = tid; e < 2 * NTOK * HD; e += 192) {
    int isK = e >> 11;
    int i = e & 2047;
    int n = i >> 5, d = i & 31;
    int iy = n >> 3, ix = n & 7;
    const float* raw = isK ? k_raw : q_raw;
    const float* dwW = isK ? pk_dw_w : pq_dw_w;
    const float* dwB = isK ? pk_dw_b : pq_dw_b;
    float s = dwB[d] + (isK ? ck : cq)[d];
    for (int di = -1; di <= 1; di++) {
      int yy = iy + di; if ((unsigned)yy >= 8u) continue;
      for (int dj = -1; dj <= 1; dj++) {
        int xx = ix + dj; if ((unsigned)xx >= 8u) continue;
        s += raw[(yy * 8 + xx) * HD + d] * dwW[d * 9 + (di + 1) * 3 + (dj + 1)];
      }
    }
    float g = raw[n * HD + d] * (1.f / (1.f + expf(-s)));
    (isK ? Kt : Qt)[n * QS + d] = (_Float16)g;
  }
  __syncthreads();

  // ---------- phase 2: attn = Q @ K^T * temp + bias ----------
  for (int t = wave; t < 16; t += 6) {
    int mt = t >> 2, nt = t & 3;
    v16h fa = load_fragA(Qt, QS, mt * 16 + ln, 0, hi);
    v16h fb = load_fragB(Kt, QS, nt * 16 + ln, 0, hi);
    f8u a;
    for (int r = 0; r < 8; r++) a.f[r] = 0.f;
    a.v = wmma32(fa, fb, a.v);
#pragma unroll
    for (int r = 0; r < 8; r++) {
      int nrow = mt * 16 + r + hi * 8;
      int mcol = nt * 16 + ln;
      int dyi = (nrow >> 3) - (mcol >> 3) + 7;
      int dxi = (nrow & 7) - (mcol & 7) + 7;
      attnS[nrow * PS + mcol] = a.f[r] * temp + biasT[(dyi * 15 + dxi) * NHEADS + head];
    }
  }
  __syncthreads();

  // ---------- softmax rows ----------
  if (tid < NTOK) {
    float* row = attnS + tid * PS;
    float m = row[0];
    for (int i = 1; i < NTOK; i++) m = fmaxf(m, row[i]);
    float s = 0.f;
    for (int i = 0; i < NTOK; i++) { float e = expf(row[i] - m); row[i] = e; s += e; }
    float inv = 1.f / s;
    _Float16* arow = At + tid * AS;
    for (int i = 0; i < NTOK; i++) arow[i] = (_Float16)(row[i] * inv);
  }
  __syncthreads();

  // ---------- phase 3: out = attn @ V ----------
  for (int t = wave; t < 8; t += 6) {
    int mt = t >> 1, nt = t & 1;
    f8u a;
    for (int r = 0; r < 8; r++) a.f[r] = 0.f;
#pragma unroll
    for (int ks2 = 0; ks2 < 2; ks2++) {
      int kb = ks2 * 32;
      v16h fa = load_fragA(At, AS, mt * 16 + ln, kb, hi);
      v16h fb = load_fragB(Vt, VS, nt * 16 + ln, kb, hi);
      a.v = wmma32(fa, fb, a.v);
    }
#pragma unroll
    for (int r = 0; r < 8; r++) {
      int token = mt * 16 + r + hi * 8;
      int d = nt * 16 + ln;
      O[((long)w * NTOK + token) * CDIM + head * HD + d] = (_Float16)a.f[r];
    }
  }
}

// ---------------------------------------------------------------------------
// K4: y = O @ Wcomb^T + b_proj, scatter to NCHW f32. LDS-free WMMA GEMM.
// grid (1024 windows, 3 n-blocks of 128), 256 threads (8 waves)
// ---------------------------------------------------------------------------
__global__ __launch_bounds__(256) void proj_kernel(
    const _Float16* __restrict__ O, const _Float16* __restrict__ Wc,
    const float* __restrict__ bproj, float* __restrict__ y)
{
  const int w = blockIdx.x;
  const int tid = threadIdx.x, wave = tid >> 5, lane = tid & 31;
  const int ln = lane & 15, hi = lane >> 4;
  const int o = blockIdx.y * 128 + wave * 16 + ln;
  const int b = w >> 8, wy = (w >> 4) & 15, wx = w & 15;

  f8u acc[4];
  for (int mt = 0; mt < 4; mt++)
    for (int r = 0; r < 8; r++) acc[mt].f[r] = 0.f;

  for (int ks = 0; ks < 12; ks++) {
    int kb = ks * 32;
    v16h fb = load_fragB(Wc, CDIM, o, kb, hi);
#pragma unroll
    for (int mt = 0; mt < 4; mt++) {
      v16h fa = load_fragA(O, CDIM, w * NTOK + mt * 16 + ln, kb, hi);
      acc[mt].v = wmma32(fa, fb, acc[mt].v);
    }
  }
  float bo = bproj[o];
#pragma unroll
  for (int mt = 0; mt < 4; mt++)
    for (int r = 0; r < 8; r++) {
      int token = mt * 16 + r + hi * 8;
      int iy = token >> 3, ix = token & 7;
      int hr = wy * 8 + iy, wc2 = wx * 8 + ix;
      y[(((long)b * CDIM + o) * 128 + hr) * 128 + wc2] = acc[mt].f[r] + bo;
    }
}

// ---------------------------------------------------------------------------
extern "C" void kernel_launch(void* const* d_in, const int* in_sizes, int n_in,
                              void* d_out, int out_size, void* d_ws, size_t ws_size,
                              hipStream_t stream) {
  (void)in_sizes; (void)n_in; (void)out_size; (void)ws_size;
  const float* x        = (const float*)d_in[0];
  const float* w_qk     = (const float*)d_in[1];
  const float* b_qk     = (const float*)d_in[2];
  const float* w_v      = (const float*)d_in[3];
  const float* b_v      = (const float*)d_in[4];
  const float* log_temp = (const float*)d_in[5];
  const float* pq_dw_w  = (const float*)d_in[6];
  const float* pq_dw_b  = (const float*)d_in[7];
  const float* pq_f1w   = (const float*)d_in[8];
  const float* pq_f1b   = (const float*)d_in[9];
  const float* pq_f2w   = (const float*)d_in[10];
  const float* pq_f2b   = (const float*)d_in[11];
  const float* pk_dw_w  = (const float*)d_in[12];
  const float* pk_dw_b  = (const float*)d_in[13];
  const float* pk_f1w   = (const float*)d_in[14];
  const float* pk_f1b   = (const float*)d_in[15];
  const float* pk_f2w   = (const float*)d_in[16];
  const float* pk_f2b   = (const float*)d_in[17];
  const float* pos_w    = (const float*)d_in[18];
  /* d_in[19] = pos_conv_b cancels in (pe_n - pe_m) */
  const float* pos_sc   = (const float*)d_in[20];
  const float* meta_w1  = (const float*)d_in[21];
  const float* meta_b1  = (const float*)d_in[22];
  const float* meta_w2  = (const float*)d_in[23];
  const float* meta_b2  = (const float*)d_in[24];
  const float* w_po     = (const float*)d_in[25];
  const float* w_proj   = (const float*)d_in[26];
  const float* b_proj   = (const float*)d_in[27];

  char* ws = (char*)d_ws;
  _Float16* WqkvH  = (_Float16*)(ws + 0);          // 1152*384*2 = 884736
  _Float16* WcombH = (_Float16*)(ws + 884736);     // 384*384*2  = 294912
  float*    bqkv   = (float*)(ws + 1179648);       // 1152*4     = 4608
  float*    biasT  = (float*)(ws + 1184256);       // 2700*4     = 10800
  _Float16* O      = (_Float16*)(ws + 1195264);    // 65536*384*2 = 50331648
  _Float16* X16    = (_Float16*)(ws + 51526912);   // 65536*384*2 = 50331648 -> 101858560

  prep_qkv_w<<<1728, 256, 0, stream>>>(w_qk, b_qk, w_v, b_v, WqkvH, bqkv);
  prep_comb_w<<<576, 256, 0, stream>>>(w_proj, w_po, WcombH);
  prep_x<<<24576, 256, 0, stream>>>(x, X16);
  bias_kernel<<<225, 128, 0, stream>>>(pos_w, pos_sc, meta_w1, meta_b1, meta_w2, meta_b2, biasT);
  attn_kernel<<<dim3(1024, NHEADS), 192, SMEM_BYTES, stream>>>(
      X16, WqkvH, bqkv, log_temp, biasT,
      pq_dw_w, pq_dw_b, pq_f1w, pq_f1b, pq_f2w, pq_f2b,
      pk_dw_w, pk_dw_b, pk_f1w, pk_f1b, pk_f2w, pk_f2b, O);
  proj_kernel<<<dim3(1024, 3), 256, 0, stream>>>(O, WcombH, b_proj, (float*)d_out);
}
